// BiLSTM_CRF_36112085024780
// MI455X (gfx1250) — compile-verified, matrix operand order checked by reading, not run on hardware
//
#include <hip/hip_runtime.h>
#include <hip/hip_bf16.h>

// ---------------------------------------------------------------------------
// BiLSTM-CRF for MI455X (gfx1250, wave32, WMMA).
// T=4096, E=256, HD=256, H=512, NT=16, START=14, STOP=15.
// ---------------------------------------------------------------------------

#define T_LEN 4096
#define E_DIM 256
#define HD_DIM 256
#define H_DIM 512
#define G_DIM 1024   // 4*HD
#define NT 16
#define NEG_INF (-10000.0f)

typedef _Float16 v8h  __attribute__((ext_vector_type(8)));
typedef _Float16 v16h __attribute__((ext_vector_type(16)));
typedef float    v8f  __attribute__((ext_vector_type(8)));

// ---------------------------------------------------------------------------
// Small conversion / packing kernels
// ---------------------------------------------------------------------------

__global__ void gather_embed_f16(const int* __restrict__ sent,
                                 const float* __restrict__ emb,
                                 _Float16* __restrict__ Xh) {
    int t = blockIdx.x;
    int e = threadIdx.x;
    Xh[(size_t)t * E_DIM + e] = (_Float16)emb[(size_t)sent[t] * E_DIM + e];
}

__global__ void f32_to_f16(const float* __restrict__ src,
                           _Float16* __restrict__ dst, int n) {
    int i = blockIdx.x * blockDim.x + threadIdx.x;
    if (i < n) dst[i] = (_Float16)src[i];
}

// dst[c][r] = src[r][c]  (dst is [cols][rows], f16)
__global__ void transpose_f32_to_f16(const float* __restrict__ src,
                                     _Float16* __restrict__ dst,
                                     int rows, int cols) {
    int c = blockIdx.x;
    for (int r = threadIdx.x; r < rows; r += blockDim.x)
        dst[(size_t)c * rows + r] = (_Float16)src[(size_t)r * cols + c];
}

__global__ void bias_combine(const float* __restrict__ bihf, const float* __restrict__ bhhf,
                             const float* __restrict__ bihb, const float* __restrict__ bhhb,
                             float* __restrict__ out) {
    int i = blockIdx.x * blockDim.x + threadIdx.x;
    if (i < G_DIM) {
        out[i]         = bihf[i] + bhhf[i];
        out[G_DIM + i] = bihb[i] + bhhb[i];
    }
}

// ---------------------------------------------------------------------------
// Generic WMMA GEMM:  C[M,N](f32) = A[M,K](f16) * Bp[N,K](f16)^T  (+epilogue)
// Bp is packed row-major [N][K] (i.e., B already "transposed").
// One wave computes one 16x16 output tile, K in chunks of 32 via
// v_wmma_f32_16x16x32_f16 with f32 accumulation.
//
// Fragment load mapping (per CDNA5 ISA 7.12.2, 16-bit A 16x32):
//   lane L: row = L&15, K-half kh = L>>4
//   halves[0..7]  = row[k0 + kh*8 .. +7]      (one 16B load)
//   halves[8..15] = row[k0 + 16 + kh*8 .. +7] (one 16B load)
// B (32x16) mirrors this with N = L&15 since Bp is stored [N][K].
//
// mode 0: C = AB + bias[n]
// mode 1: C = tanh(AB)
// mode 2: C = alpha[m]*(AB) + bias[n]
// ---------------------------------------------------------------------------

__device__ __forceinline__ v16h frag_load(const _Float16* __restrict__ row,
                                          int k0, int kh) {
    v8h lo = *(const v8h*)(row + k0 + kh * 8);
    v8h hi = *(const v8h*)(row + k0 + 16 + kh * 8);
    v16h r;
#pragma unroll
    for (int i = 0; i < 8; ++i) { r[i] = lo[i]; r[8 + i] = hi[i]; }
    return r;
}

__global__ __launch_bounds__(256)
void wmma_gemm(const _Float16* __restrict__ A, int lda,
               const _Float16* __restrict__ Bp, int ldb,
               float* __restrict__ C, int ldc,
               int Mtiles, int Ntiles, int K,
               const float* __restrict__ bias,
               const float* __restrict__ alpha,
               int mode) {
    int wave = threadIdx.x >> 5;
    int lane = threadIdx.x & 31;
    int tile = blockIdx.x * (blockDim.x >> 5) + wave;
    int total = Mtiles * Ntiles;
    if (tile >= total) return;

    int mt = tile / Ntiles, nt = tile % Ntiles;
    int m0 = mt * 16, n0 = nt * 16;
    int r  = lane & 15;   // A row within tile / B "row" (output col) within tile
    int kh = lane >> 4;   // K-half select

    const _Float16* Arow = A  + (size_t)(m0 + r) * lda;
    const _Float16* Brow = Bp + (size_t)(n0 + r) * ldb;

    v8f c = {};
    for (int k0 = 0; k0 < K; k0 += 32) {
        v16h a = frag_load(Arow, k0, kh);
        v16h b = frag_load(Brow, k0, kh);
        // D = A*B + C, f32 accumulate (v_wmma_f32_16x16x32_f16)
        c = __builtin_amdgcn_wmma_f32_16x16x32_f16(
                /*neg_a=*/false, a, /*neg_b=*/false, b,
                /*c_mod=*/(short)0, c, /*reuse_a=*/false, /*reuse_b=*/false);
    }

    // C/D layout: VGPR rr -> M = rr + kh*8 ; N = lane&15
    int n = lane & 15;
#pragma unroll
    for (int rr = 0; rr < 8; ++rr) {
        int m = m0 + rr + kh * 8;
        float v = c[rr];
        if (mode == 0)      v += bias[n0 + n];
        else if (mode == 1) v = tanhf(v);
        else                v = alpha[m] * v + bias[n0 + n];
        C[(size_t)m * ldc + n0 + n] = v;
    }
}

// ---------------------------------------------------------------------------
// Sequential LSTM recurrence. One workgroup per direction (grid = 2), the
// two directions execute concurrently on separate WGPs.
// 512 threads; thread j owns gate rows j and j+512 of Whh (f16), kept
// register-resident (2 x 32 x v8h = 256 VGPRs of weights). Per step:
//   g[j] = pregate[t][j] + Whh[j,:] . h   (h broadcast from LDS)
// then threads u<256 apply the cell update.
// ---------------------------------------------------------------------------

__device__ __forceinline__ float sigm(float x) {
    return 1.0f / (1.0f + __expf(-x));
}

__global__ __launch_bounds__(512)
void lstm_recur(const _Float16* __restrict__ Whh_h,   // [2][1024][256] f16
                const float* __restrict__ pregate,    // [2][T][1024] f32
                const float* __restrict__ h0,         // [2][1][256]
                const float* __restrict__ c0,         // [2][1][256]
                float* __restrict__ out_f32,          // [T][512]
                _Float16* __restrict__ out_h) {       // [T][512] f16
    int dir = blockIdx.x;
    int j   = threadIdx.x;                 // 0..511
    __shared__ float hs[HD_DIM];
    __shared__ float cs[HD_DIM];
    __shared__ float g4[G_DIM];

    if (j < HD_DIM) {
        hs[j] = h0[dir * HD_DIM + j];
        cs[j] = c0[dir * HD_DIM + j];
    }

    // Register-resident recurrent weights for rows j and j+512.
    const _Float16* w0 = Whh_h + ((size_t)dir * G_DIM + j) * HD_DIM;
    const _Float16* w1 = Whh_h + ((size_t)dir * G_DIM + j + 512) * HD_DIM;
    v8h wr0[32], wr1[32];
#pragma unroll
    for (int k = 0; k < 32; ++k) { wr0[k] = *(const v8h*)(w0 + k * 8); }
#pragma unroll
    for (int k = 0; k < 32; ++k) { wr1[k] = *(const v8h*)(w1 + k * 8); }

    const float* pg = pregate + (size_t)dir * T_LEN * G_DIM;
    __syncthreads();

    for (int s = 0; s < T_LEN; ++s) {
        int t = dir ? (T_LEN - 1 - s) : s;
        float acc0 = pg[(size_t)t * G_DIM + j];
        float acc1 = pg[(size_t)t * G_DIM + j + 512];
        if (s + 1 < T_LEN) {
            int tn = dir ? (t - 1) : (t + 1);
            __builtin_prefetch(pg + (size_t)tn * G_DIM + j, 0, 3);
            __builtin_prefetch(pg + (size_t)tn * G_DIM + j + 512, 0, 3);
        }
#pragma unroll 4
        for (int k = 0; k < 32; ++k) {
            v8h a = wr0[k], b = wr1[k];
#pragma unroll
            for (int i = 0; i < 8; ++i) {
                float hv = hs[k * 8 + i];
                acc0 += (float)a[i] * hv;   // lowers to v_fma_mix
                acc1 += (float)b[i] * hv;
            }
        }
        g4[j]       = acc0;
        g4[j + 512] = acc1;
        __syncthreads();
        if (j < HD_DIM) {
            float ig = sigm(g4[j]);
            float fg = sigm(g4[HD_DIM + j]);
            float gg = tanhf(g4[2 * HD_DIM + j]);
            float og = sigm(g4[3 * HD_DIM + j]);
            float c  = fg * cs[j] + ig * gg;
            float h  = og * tanhf(c);
            cs[j] = c; hs[j] = h;
            size_t o = (size_t)t * H_DIM + dir * HD_DIM + j;
            out_f32[o] = h;
            out_h[o]   = (_Float16)h;
        }
        __syncthreads();
    }
}

// ---------------------------------------------------------------------------
// logits[t] = tanhmat[t,:] . u_omega   (one wave per t)
// ---------------------------------------------------------------------------
__global__ __launch_bounds__(256)
void logits_dot(const float* __restrict__ tanhmat,
                const float* __restrict__ u,
                float* __restrict__ logits) {
    int wave = threadIdx.x >> 5, lane = threadIdx.x & 31;
    int t = blockIdx.x * 8 + wave;
    const float* row = tanhmat + (size_t)t * H_DIM;
    float s = 0.f;
    for (int i = lane; i < H_DIM; i += 32) s += row[i] * u[i];
#pragma unroll
    for (int off = 16; off; off >>= 1) s += __shfl_xor(s, off);
    if (lane == 0) logits[t] = s;
}

// ---------------------------------------------------------------------------
// softmax over the 4096 logits (single block, 32 waves)
// ---------------------------------------------------------------------------
__global__ __launch_bounds__(1024)
void softmax4096(const float* __restrict__ logits, float* __restrict__ alphas) {
    __shared__ float red[32];
    __shared__ float stat[2];
    int tid = threadIdx.x, lane = tid & 31, wv = tid >> 5;

    float m = -3.4e38f;
    for (int i = tid; i < T_LEN; i += 1024) m = fmaxf(m, logits[i]);
#pragma unroll
    for (int off = 16; off; off >>= 1) m = fmaxf(m, __shfl_xor(m, off));
    if (lane == 0) red[wv] = m;
    __syncthreads();
    if (wv == 0) {
        float v = red[lane];
#pragma unroll
        for (int off = 16; off; off >>= 1) v = fmaxf(v, __shfl_xor(v, off));
        if (lane == 0) stat[0] = v;
    }
    __syncthreads();
    float mx = stat[0];

    float s = 0.f;
    for (int i = tid; i < T_LEN; i += 1024) s += __expf(logits[i] - mx);
#pragma unroll
    for (int off = 16; off; off >>= 1) s += __shfl_xor(s, off);
    if (lane == 0) red[wv] = s;
    __syncthreads();
    if (wv == 0) {
        float v = red[lane];
#pragma unroll
        for (int off = 16; off; off >>= 1) v += __shfl_xor(v, off);
        if (lane == 0) stat[1] = v;
    }
    __syncthreads();
    float inv = 1.0f / stat[1];
    for (int i = tid; i < T_LEN; i += 1024)
        alphas[i] = __expf(logits[i] - mx) * inv;
}

// ---------------------------------------------------------------------------
// Viterbi: forward scan (16x16 per step in one block) + backtrace.
// out[0] = score, out[1..T] = path (as float).
// ---------------------------------------------------------------------------
__global__ __launch_bounds__(256)
void viterbi(const float* __restrict__ feats,   // [T][16]
             const float* __restrict__ trans,   // [16][16]  trans[next][prev]
             int* __restrict__ bps,             // [T][16]
             float* __restrict__ out) {
    __shared__ float tr[256], sm[256], fv[NT], fvn[NT];
    int tid = threadIdx.x;
    tr[tid] = trans[tid];
    if (tid < NT) fv[tid] = (tid == 14) ? 0.0f : NEG_INF;  // START=14
    __syncthreads();

    int n = tid >> 4, p = tid & 15;
    for (int t = 0; t < T_LEN; ++t) {
        sm[tid] = fv[p] + tr[n * NT + p];
        __syncthreads();
        if (tid < NT) {
            float best = sm[tid * NT]; int bi = 0;
#pragma unroll
            for (int q = 1; q < NT; ++q) {
                float s = sm[tid * NT + q];
                if (s > best) { best = s; bi = q; }
            }
            fvn[tid] = best + feats[(size_t)t * NT + tid];
            bps[(size_t)t * NT + tid] = bi;
        }
        __syncthreads();
        if (tid < NT) fv[tid] = fvn[tid];
        __syncthreads();
    }

    if (tid == 0) {
        // terminal = fv + trans[STOP=15][:]
        float best = fv[0] + tr[15 * NT + 0]; int bi = 0;
        for (int q = 1; q < NT; ++q) {
            float s = fv[q] + tr[15 * NT + q];
            if (s > best) { best = s; bi = q; }
        }
        out[0] = best;
        int tag = bi;
        for (int t = T_LEN - 1; t >= 0; --t) {
            out[1 + t] = (float)tag;
            tag = bps[(size_t)t * NT + tag];
        }
    }
}

// ---------------------------------------------------------------------------
// Host-side orchestration
// ---------------------------------------------------------------------------
extern "C" void kernel_launch(void* const* d_in, const int* in_sizes, int n_in,
                              void* d_out, int out_size, void* d_ws, size_t ws_size,
                              hipStream_t stream) {
    (void)in_sizes; (void)n_in; (void)out_size; (void)ws_size;

    const int*   sentence = (const int*)  d_in[0];
    const float* emb      = (const float*)d_in[1];
    const float* Wih_f    = (const float*)d_in[2];
    const float* Whh_f    = (const float*)d_in[3];
    const float* bih_f    = (const float*)d_in[4];
    const float* bhh_f    = (const float*)d_in[5];
    const float* Wih_b    = (const float*)d_in[6];
    const float* Whh_b    = (const float*)d_in[7];
    const float* bih_b    = (const float*)d_in[8];
    const float* bhh_b    = (const float*)d_in[9];
    const float* h0       = (const float*)d_in[10];
    const float* c0       = (const float*)d_in[11];
    const float* w_omega  = (const float*)d_in[12];
    const float* u_omega  = (const float*)d_in[13];
    const float* W_tag    = (const float*)d_in[14];
    const float* b_tag    = (const float*)d_in[15];
    const float* trans    = (const float*)d_in[16];

    // Workspace carving (256B aligned regions)
    char* wp = (char*)d_ws;
    auto carve = [&](size_t bytes) -> void* {
        void* r = (void*)wp;
        wp += (bytes + 255) & ~(size_t)255;
        return r;
    };
    _Float16* Xh      = (_Float16*)carve((size_t)T_LEN * E_DIM * 2);        // 2 MB
    _Float16* Wih_h   = (_Float16*)carve((size_t)2 * G_DIM * E_DIM * 2);    // 1 MB
    _Float16* Whh_h   = (_Float16*)carve((size_t)2 * G_DIM * HD_DIM * 2);   // 1 MB
    float*    biasc   = (float*)   carve((size_t)2 * G_DIM * 4);            // 8 KB
    float*    pregate = (float*)   carve((size_t)2 * T_LEN * G_DIM * 4);    // 32 MB
    float*    out_f32 = (float*)   carve((size_t)T_LEN * H_DIM * 4);        // 8 MB
    _Float16* out_h   = (_Float16*)carve((size_t)T_LEN * H_DIM * 2);        // 4 MB
    _Float16* wT_h    = (_Float16*)carve((size_t)H_DIM * H_DIM * 2);        // 0.5 MB
    _Float16* Wtag_h  = (_Float16*)carve((size_t)NT * H_DIM * 2);           // 16 KB
    float*    logits  = (float*)   carve((size_t)T_LEN * 4);
    float*    alphas  = (float*)   carve((size_t)T_LEN * 4);
    float*    feats   = (float*)   carve((size_t)T_LEN * NT * 4);
    int*      bps     = (int*)     carve((size_t)T_LEN * NT * 4);
    float*    tanhmat = pregate;   // reuse: pregate dead after recurrence

    // 1. Embedding gather -> f16
    gather_embed_f16<<<T_LEN, E_DIM, 0, stream>>>(sentence, emb, Xh);

    // 2. Weight conversions (f32 -> f16)
    const int WN = G_DIM * E_DIM;  // 262144
    f32_to_f16<<<(WN + 255) / 256, 256, 0, stream>>>(Wih_f, Wih_h,        WN);
    f32_to_f16<<<(WN + 255) / 256, 256, 0, stream>>>(Wih_b, Wih_h + WN,   WN);
    f32_to_f16<<<(WN + 255) / 256, 256, 0, stream>>>(Whh_f, Whh_h,        WN);
    f32_to_f16<<<(WN + 255) / 256, 256, 0, stream>>>(Whh_b, Whh_h + WN,   WN);
    bias_combine<<<(G_DIM + 255) / 256, 256, 0, stream>>>(bih_f, bhh_f, bih_b, bhh_b, biasc);
    transpose_f32_to_f16<<<H_DIM, 256, 0, stream>>>(w_omega, wT_h, H_DIM, H_DIM);
    f32_to_f16<<<(NT * H_DIM + 255) / 256, 256, 0, stream>>>(W_tag, Wtag_h, NT * H_DIM);

    // 3. Input projections (WMMA): pregate[dir] = X @ Wih^T + (bih + bhh)
    {
        int Mtiles = T_LEN / 16, Ntiles = G_DIM / 16;       // 256 x 64
        int tiles = Mtiles * Ntiles, grid = (tiles + 7) / 8;
        wmma_gemm<<<grid, 256, 0, stream>>>(Xh, E_DIM, Wih_h, E_DIM,
                                            pregate, G_DIM,
                                            Mtiles, Ntiles, E_DIM,
                                            biasc, nullptr, 0);
        wmma_gemm<<<grid, 256, 0, stream>>>(Xh, E_DIM, Wih_h + WN, E_DIM,
                                            pregate + (size_t)T_LEN * G_DIM, G_DIM,
                                            Mtiles, Ntiles, E_DIM,
                                            biasc + G_DIM, nullptr, 0);
    }

    // 4. Sequential bidirectional recurrence (2 concurrent workgroups)
    lstm_recur<<<2, 512, 0, stream>>>(Whh_h, pregate, h0, c0, out_f32, out_h);

    // 5. Attention: tanhmat = tanh(out @ w_omega)   (WMMA)
    {
        int Mtiles = T_LEN / 16, Ntiles = H_DIM / 16;       // 256 x 32
        int tiles = Mtiles * Ntiles, grid = (tiles + 7) / 8;
        wmma_gemm<<<grid, 256, 0, stream>>>(out_h, H_DIM, wT_h, H_DIM,
                                            tanhmat, H_DIM,
                                            Mtiles, Ntiles, H_DIM,
                                            nullptr, nullptr, 1);
    }

    // 6. logits, softmax
    logits_dot<<<T_LEN / 8, 256, 0, stream>>>(tanhmat, u_omega, logits);
    softmax4096<<<1, 1024, 0, stream>>>(logits, alphas);

    // 7. feats = alpha[t] * (out @ W_tag^T) + b_tag   (WMMA, N=16 -> 1 tile col)
    {
        int Mtiles = T_LEN / 16, Ntiles = 1;
        int tiles = Mtiles * Ntiles, grid = (tiles + 7) / 8;
        wmma_gemm<<<grid, 256, 0, stream>>>(out_h, H_DIM, Wtag_h, H_DIM,
                                            feats, NT,
                                            Mtiles, Ntiles, H_DIM,
                                            b_tag, alphas, 2);
    }

    // 8. Viterbi decode -> d_out = [score, path...]
    viterbi<<<1, 256, 0, stream>>>(feats, trans, bps, (float*)d_out);
}
